// CDDKernel_46703474377361
// MI455X (gfx1250) — matrix operand order repaired and checked by reading, not compiled, not run on hardware
//
#include <hip/hip_runtime.h>
#include <math.h>

// ---------------- problem constants ----------------
#define NC 12      // classes
#define NN 384     // samples per side
#define ND 256     // feature dim
#define NT (NN*NN) // 147456

typedef __attribute__((ext_vector_type(16))) __bf16 v16bf;
typedef __attribute__((ext_vector_type(8)))  float  v8f;

// ---------------- workspace layout (bytes) ----------------
constexpr size_t OFF_ESS   = 0;                          // f32 [384*384] * 3 (ss,tt,st contiguous)
constexpr size_t OFF_BFS   = OFF_ESS   + 3ull*NT*4;      // u16 [384*256]
constexpr size_t OFF_BFT   = OFF_BFS   + (size_t)NN*ND*2;
constexpr size_t OFF_NORMS = OFF_BFT   + (size_t)NN*ND*2; // f32 [768]
constexpr size_t OFF_CSF   = OFF_NORMS + 768*4;          // f32 [12]
constexpr size_t OFF_CT    = OFF_CSF   + 12*4;           // f32 [12]
constexpr size_t OFF_SSTS  = OFF_CT    + 12*4;           // f32 [144]
constexpr size_t OFF_SST   = OFF_SSTS  + 144*4;          // f32 [144]
constexpr size_t OFF_STT   = OFF_SST   + 144*4;          // f32 [12]
constexpr size_t OFF_GIN   = OFF_STT   + 12*4;           // f32 [12]
constexpr size_t OFF_G2    = OFF_GIN   + 12*4;           // f32 [144]
constexpr size_t OFF_K1    = OFF_G2    + 144*4;          // f32 [12]
constexpr size_t OFF_K3    = OFF_K1    + 12*4;           // f32 [12]
constexpr size_t OFF_K2P   = OFF_K3    + 12*4;           // f32 [144]
constexpr size_t OFF_MMD   = OFF_K2P   + 144*4;          // f32 [144]
constexpr size_t OFF_WTT   = OFF_MMD   + 144*4;          // f32 [384*12]
constexpr size_t OFF_WST   = OFF_WTT   + (size_t)NN*NC*4;
constexpr size_t OFF_CLOFF = OFF_WST   + (size_t)NN*NC*4; // i32 [16]
constexpr size_t OFF_ORDER = OFF_CLOFF + 16*4;            // i32 [384]

__device__ __forceinline__ unsigned short f32_to_bf16_rne(float f) {
    unsigned int u = __float_as_uint(f);
    unsigned int r = u + 0x7FFFu + ((u >> 16) & 1u);
    return (unsigned short)(r >> 16);
}

// multi-bandwidth RBF: sum_i exp(-clip(d / max(g*2^(i-2),1e-5), 1e-5, 1e5))
__device__ __forceinline__ float kern5(float d, float g) {
    float acc = 0.0f;
    float m = 0.25f;
#pragma unroll
    for (int i = 0; i < 5; ++i) {
        float bw = fmaxf(g * m, 1e-5f);
        float r  = fminf(fmaxf(d / bw, 1e-5f), 1e5f);
        acc += __expf(-r);
        m *= 2.0f;
    }
    return acc;
}

// ---------------- K0: f32 -> bf16 convert + row sq-norms ----------------
__global__ void cdd_convert_norms(const float* __restrict__ src,
                                  const float* __restrict__ tgt,
                                  unsigned short* __restrict__ bfS,
                                  unsigned short* __restrict__ bfT,
                                  float* __restrict__ norms) {
    int row = blockIdx.x;               // 0..767
    int tid = threadIdx.x;              // 256 = ND
    const float* x = (row < NN) ? src + (size_t)row * ND : tgt + (size_t)(row - NN) * ND;
    unsigned short* o = (row < NN) ? bfS + (size_t)row * ND : bfT + (size_t)(row - NN) * ND;
    float v = x[tid];
    o[tid] = f32_to_bf16_rne(v);
    __shared__ float red[256];
    red[tid] = v * v;
    __syncthreads();
    for (int s = 128; s > 0; s >>= 1) {
        if (tid < s) red[tid] += red[tid + s];
        __syncthreads();
    }
    if (tid == 0) norms[row] = red[0];
}

// ---------------- K1: class lists (deterministic), cs, ct ----------------
__global__ void cdd_class_setup(const int* __restrict__ sy,
                                const float* __restrict__ pt,
                                float* __restrict__ csf, float* __restrict__ ct,
                                int* __restrict__ cls_off, int* __restrict__ order) {
    int tid = threadIdx.x;
    if (tid == 0) {
        int cnt[NC];
        for (int c = 0; c < NC; ++c) cnt[c] = 0;
        for (int i = 0; i < NN; ++i) cnt[sy[i]]++;
        int off = 0;
        for (int c = 0; c < NC; ++c) { cls_off[c] = off; off += cnt[c]; csf[c] = (float)cnt[c]; }
        cls_off[NC] = off;
        int pos[NC];
        for (int c = 0; c < NC; ++c) pos[c] = cls_off[c];
        for (int i = 0; i < NN; ++i) { int c = sy[i]; order[pos[c]++] = i; }
    }
    if (tid >= 1 && tid <= NC) {
        int c = tid - 1;
        float s = 0.0f;
        for (int i = 0; i < NN; ++i) s += pt[i * NC + c];
        ct[c] = s;
    }
}

// ---------------- K2: pairwise distances via WMMA bf16 Gram ----------------
// one wave -> one 16x16 tile of one of {E_ss, E_tt, E_st}
__global__ void cdd_dist_wmma(const unsigned short* __restrict__ bfS_,
                              const unsigned short* __restrict__ bfT_,
                              const float* __restrict__ norms,
                              float* __restrict__ Ebase) {
    const __bf16* bfS = reinterpret_cast<const __bf16*>(bfS_);
    const __bf16* bfT = reinterpret_cast<const __bf16*>(bfT_);
    int wave = blockIdx.x * (blockDim.x >> 5) + (threadIdx.x >> 5);
    int lane = threadIdx.x & 31;
    int m  = wave / 576;                 // 0: ss, 1: tt, 2: st
    int t  = wave % 576;
    int bi = t / 24, bj = t % 24;
    const __bf16* A = (m == 1) ? bfT : bfS;
    const __bf16* B = (m == 0) ? bfS : bfT;
    const float*  na = (m == 1) ? norms + NN : norms;
    const float*  nb = (m == 0) ? norms      : norms + NN;
    float* Eo = Ebase + (size_t)m * NT;

    int half = lane >> 4, lr = lane & 15;
    const __bf16* pa = A + (size_t)(bi * 16 + lr) * ND;  // A rows (M dimension)
    const __bf16* pb = B + (size_t)(bj * 16 + lr) * ND;  // B cols = rows of other matrix

    v8f acc = {};
#pragma unroll
    for (int k0 = 0; k0 < ND; k0 += 32) {
        // ISA 16-bit A 16x32 layout: lanes 0-15 hold K 0-7 / 16-23, lanes 16-31 hold K 8-15 / 24-31
        const __bf16* qa = pa + k0 + half * 8;
        // B uses same K order as A: lanes 0-15 hold K 0-7 / 16-23, lanes 16-31 hold K 8-15 / 24-31
        const __bf16* qb = pb + k0 + half * 8;
        v16bf a, b;
#pragma unroll
        for (int e = 0; e < 8; ++e) { a[e] = qa[e]; a[8 + e] = qa[16 + e]; }
#pragma unroll
        for (int e = 0; e < 8; ++e) { b[e] = qb[e]; b[8 + e] = qb[16 + e]; }
        acc = __builtin_amdgcn_wmma_f32_16x16x32_bf16(false, a, false, b,
                                                      (short)0, acc, false, false);
    }

    int j = bj * 16 + lr;
    float nbj = nb[j];
#pragma unroll
    for (int v = 0; v < 8; ++v) {         // C layout: VGPR v -> M = v + half*8, N = lane%16
        int i = bi * 16 + v + half * 8;
        float d2 = na[i] + nbj - 2.0f * acc[v];
        Eo[(size_t)i * NN + j] = sqrtf(fmaxf(d2, 0.0f));
    }
}

// ---------------- K3a: W_tt = E_tt @ pt, W_st = E_st @ pt ----------------
__global__ void cdd_matvec_w(const float* __restrict__ Ett,
                             const float* __restrict__ Est,
                             const float* __restrict__ pt,
                             float* __restrict__ Wtt, float* __restrict__ Wst) {
    int g = blockIdx.x * blockDim.x + threadIdx.x;   // 0..9215
    if (g >= 2 * NN * NC) return;
    int which = g / (NN * NC);
    int r = g % (NN * NC);
    int i = r / NC, c = r % NC;
    const float* E = which ? Est : Ett;
    float s = 0.0f;
    for (int j = 0; j < NN; ++j) s += E[(size_t)i * NN + j] * pt[j * NC + c];
    (which ? Wst : Wtt)[i * NC + c] = s;
}

// ---------------- K3b: class stats -> g_in, g2 ----------------
__global__ void cdd_stats(const float* __restrict__ Ess, const float* __restrict__ pt,
                          const float* __restrict__ Wtt, const float* __restrict__ Wst,
                          const float* __restrict__ csf, const float* __restrict__ ct,
                          const int* __restrict__ cls_off, const int* __restrict__ order,
                          float* __restrict__ sst_s, float* __restrict__ sst,
                          float* __restrict__ stt, float* __restrict__ g_in,
                          float* __restrict__ g2) {
    int tid = threadIdx.x;   // 256
    if (tid < 144) {
        int s = tid / NC, t = tid % NC;
        float a = 0.0f, b = 0.0f;
        for (int ii = cls_off[s]; ii < cls_off[s + 1]; ++ii) {
            int i = order[ii];
            const float* row = Ess + (size_t)i * NN;
            float rs = 0.0f;
            for (int jj = cls_off[t]; jj < cls_off[t + 1]; ++jj) rs += row[order[jj]];
            a += rs;
            b += Wst[i * NC + t];
        }
        sst_s[tid] = a;   // sum_{i in s, j in t} E_ss[i,j]
        sst[tid]   = b;   // sum_{i in s} sum_j E_st[i,j] pt[j,t]
    } else if (tid < 156) {
        int c = tid - 144;
        float s = 0.0f;
        for (int i = 0; i < NN; ++i) s += pt[i * NC + c] * Wtt[i * NC + c];
        stt[c] = s;
    }
    __syncthreads();
    if (tid < NC) {
        int c = tid;
        float cs = csf[c], cc = ct[c];
        g_in[c] = (sst_s[c * NC + c] + stt[c] + 2.0f * sst[c * NC + c]) /
                  (cs * cs + cc * cc + 2.0f * cs * cc);
    }
    if (tid < 144) {
        int s = tid / NC, t = tid % NC;
        float a = csf[s], b = csf[t];
        g2[tid] = (sst_s[s * NC + s] + sst_s[t * NC + t] + 2.0f * sst_s[tid]) /
                  (a * a + b * b + 2.0f * a * b);
    }
}

// ---------------- K4a: intra structured terms k1[c], k3[c] ----------------
__global__ void cdd_intra_struct(const float* __restrict__ Ess, const float* __restrict__ Est,
                                 const float* __restrict__ pt, const float* __restrict__ g_in,
                                 const float* __restrict__ csf, const float* __restrict__ ct,
                                 const int* __restrict__ cls_off, const int* __restrict__ order,
                                 float* __restrict__ k1b, float* __restrict__ k3b) {
    int c = blockIdx.x, tid = threadIdx.x;
    float g = g_in[c];
    int o0 = cls_off[c], n = cls_off[c + 1] - o0;
    float s1 = 0.0f, s3 = 0.0f;
    int tot1 = n * n;
    for (int p = tid; p < tot1; p += 256) {
        int i = order[o0 + p / n], j = order[o0 + p % n];
        s1 += kern5(Ess[(size_t)i * NN + j], g);
    }
    int tot3 = n * NN;
    for (int p = tid; p < tot3; p += 256) {
        int i = order[o0 + p / NN], j = p % NN;
        float w = pt[j * NC + c];
        s3 += kern5(Est[(size_t)i * NN + j] * w, g) * w;
    }
    __shared__ float r1[256], r3[256];
    r1[tid] = s1; r3[tid] = s3;
    __syncthreads();
    for (int s = 128; s > 0; s >>= 1) {
        if (tid < s) { r1[tid] += r1[tid + s]; r3[tid] += r3[tid + s]; }
        __syncthreads();
    }
    if (tid == 0) {
        float cs = csf[c];
        k1b[c] = r1[0] / (cs * cs);
        k3b[c] = r3[0] / (cs * ct[c]);
    }
}

// ---------------- K4b: intra dense term k2 (soft pt on both sides) ----------------
__global__ void cdd_intra_dense(const float* __restrict__ Ett, const float* __restrict__ pt,
                                const float* __restrict__ g_in, float* __restrict__ k2part) {
    int c = blockIdx.x / 12, rb = blockIdx.x % 12;  // 12 row-chunks of 32
    int tid = threadIdx.x;
    float g = g_in[c];
    float s = 0.0f;
    int base = rb * 32;
    for (int p = tid; p < 32 * NN; p += 256) {
        int i = base + p / NN, j = p % NN;
        float w = pt[i * NC + c] * pt[j * NC + c];
        s += kern5(Ett[(size_t)i * NN + j] * w, g) * w;
    }
    __shared__ float r[256];
    r[tid] = s;
    __syncthreads();
    for (int st = 128; st > 0; st >>= 1) {
        if (tid < st) r[tid] += r[tid + st];
        __syncthreads();
    }
    if (tid == 0) k2part[blockIdx.x] = r[0];
}

// ---------------- K4c: inter class-pair MMD ----------------
__global__ void cdd_inter(const float* __restrict__ Ess, const float* __restrict__ g2,
                          const float* __restrict__ csf, const int* __restrict__ cls_off,
                          const int* __restrict__ order, float* __restrict__ mmd) {
    int s = blockIdx.x / NC, t = blockIdx.x % NC;
    int tid = threadIdx.x;
    if (s == t) { if (tid == 0) mmd[blockIdx.x] = 0.0f; return; }
    float g = g2[blockIdx.x];
    int os = cls_off[s], ns = cls_off[s + 1] - os;
    int ot = cls_off[t], nt = cls_off[t + 1] - ot;
    float a1 = 0.0f, a2 = 0.0f, a3 = 0.0f;
    for (int p = tid; p < ns * ns; p += 256) {
        int i = order[os + p / ns], j = order[os + p % ns];
        a1 += kern5(Ess[(size_t)i * NN + j], g);
    }
    for (int p = tid; p < nt * nt; p += 256) {
        int i = order[ot + p / nt], j = order[ot + p % nt];
        a2 += kern5(Ess[(size_t)i * NN + j], g);
    }
    for (int p = tid; p < ns * nt; p += 256) {
        int i = order[os + p / nt], j = order[ot + p % nt];
        a3 += kern5(Ess[(size_t)i * NN + j], g);
    }
    __shared__ float r1[256], r2[256], r3[256];
    r1[tid] = a1; r2[tid] = a2; r3[tid] = a3;
    __syncthreads();
    for (int st = 128; st > 0; st >>= 1) {
        if (tid < st) { r1[tid] += r1[tid + st]; r2[tid] += r2[tid + st]; r3[tid] += r3[tid + st]; }
        __syncthreads();
    }
    if (tid == 0) {
        float a = csf[s], b = csf[t];
        mmd[blockIdx.x] = r1[0] / (a * a) + r2[0] / (b * b) - 2.0f * r3[0] / (a * b);
    }
}

// ---------------- K5: finalize ----------------
__global__ void cdd_finalize(const float* __restrict__ k1b, const float* __restrict__ k2part,
                             const float* __restrict__ k3b, const float* __restrict__ ct,
                             const float* __restrict__ mmd, float* __restrict__ out) {
    if (threadIdx.x == 0) {
        float intra = 0.0f;
        for (int c = 0; c < NC; ++c) {
            float k2 = 0.0f;
            for (int rb = 0; rb < 12; ++rb) k2 += k2part[c * 12 + rb];
            k2 /= ct[c] * ct[c];
            intra += k1b[c] + k2 - 2.0f * k3b[c];
        }
        intra /= (float)NC;
        float inter = 0.0f;
        for (int i = 0; i < NC * NC; ++i) inter += mmd[i];
        inter /= (float)(NC * NC - NC);
        out[0] = intra;
        out[1] = inter;
    }
}

extern "C" void kernel_launch(void* const* d_in, const int* in_sizes, int n_in,
                              void* d_out, int out_size, void* d_ws, size_t ws_size,
                              hipStream_t stream) {
    const float* src_x = (const float*)d_in[0];   // 384x256 f32
    const float* tgt_x = (const float*)d_in[1];   // 384x256 f32
    const int*   src_y = (const int*)d_in[2];     // 384 i32
    const float* tgt_y = (const float*)d_in[3];   // 384x12 f32
    float* out = (float*)d_out;                   // [intra, inter]

    char* ws = (char*)d_ws;
    float*          Ess   = (float*)(ws + OFF_ESS);
    float*          Ett   = Ess + NT;
    float*          Est   = Ett + NT;
    unsigned short* bfS   = (unsigned short*)(ws + OFF_BFS);
    unsigned short* bfT   = (unsigned short*)(ws + OFF_BFT);
    float*          norms = (float*)(ws + OFF_NORMS);
    float*          csf   = (float*)(ws + OFF_CSF);
    float*          ct    = (float*)(ws + OFF_CT);
    float*          sst_s = (float*)(ws + OFF_SSTS);
    float*          sst   = (float*)(ws + OFF_SST);
    float*          stt   = (float*)(ws + OFF_STT);
    float*          g_in  = (float*)(ws + OFF_GIN);
    float*          g2    = (float*)(ws + OFF_G2);
    float*          k1b   = (float*)(ws + OFF_K1);
    float*          k3b   = (float*)(ws + OFF_K3);
    float*          k2p   = (float*)(ws + OFF_K2P);
    float*          mmd   = (float*)(ws + OFF_MMD);
    float*          Wtt   = (float*)(ws + OFF_WTT);
    float*          Wst   = (float*)(ws + OFF_WST);
    int*            cloff = (int*)(ws + OFF_CLOFF);
    int*            order = (int*)(ws + OFF_ORDER);

    cdd_convert_norms<<<2 * NN, ND, 0, stream>>>(src_x, tgt_x, bfS, bfT, norms);
    cdd_class_setup<<<1, 64, 0, stream>>>(src_y, tgt_y, csf, ct, cloff, order);
    // 3 matrices x 24x24 tiles = 1728 waves, 4 waves/block
    cdd_dist_wmma<<<432, 128, 0, stream>>>(bfS, bfT, norms, Ess);
    cdd_matvec_w<<<36, 256, 0, stream>>>(Ett, Est, tgt_y, Wtt, Wst);
    cdd_stats<<<1, 256, 0, stream>>>(Ess, tgt_y, Wtt, Wst, csf, ct, cloff, order,
                                     sst_s, sst, stt, g_in, g2);
    cdd_intra_struct<<<NC, 256, 0, stream>>>(Ess, Est, tgt_y, g_in, csf, ct, cloff, order, k1b, k3b);
    cdd_intra_dense<<<NC * 12, 256, 0, stream>>>(Ett, tgt_y, g_in, k2p);
    cdd_inter<<<NC * NC, 256, 0, stream>>>(Ess, g2, csf, cloff, order, mmd);
    cdd_finalize<<<1, 32, 0, stream>>>(k1b, k2p, k3b, ct, mmd, out);
}
